// RNN_41489384080027
// MI455X (gfx1250) — matrix operand order repaired and checked by reading, not compiled, not run on hardware
//
#include <hip/hip_runtime.h>
#include <stdint.h>

typedef __attribute__((ext_vector_type(16))) _Float16 v16h;
typedef __attribute__((ext_vector_type(8)))  _Float16 v8h;
typedef __attribute__((ext_vector_type(8)))  float    v8f;
typedef __attribute__((ext_vector_type(4)))  float    v4f;

#define T_SEQ 64
#define NBATCH 16384
#define HID 20
#define VOC 29
#define WAVES_PER_BLOCK 8
#define ROWS_PER_WAVE 16
#define XSTRIDE 480   // per-wave stage size in floats (464 payload + dump/pad), 16B mult

#if __has_builtin(__builtin_amdgcn_s_wait_asynccnt)
#define WAIT_ASYNC(n) __builtin_amdgcn_s_wait_asynccnt(n)
#else
#define WAIT_ASYNC(n) asm volatile("s_wait_asynccnt %0" :: "i"(n) : "memory")
#endif

// 8x hardware tanh (gfx1250 V_TANH_F32) in one block: each tanh is the
// "independent op" covering the previous one; single trailing v_nop covers the last.
__device__ __forceinline__ void hw_tanh8(v8f& v) {
    float a0 = v[0], a1 = v[1], a2 = v[2], a3 = v[3];
    float a4 = v[4], a5 = v[5], a6 = v[6], a7 = v[7];
    asm volatile(
        "v_tanh_f32 %0, %0\n\t"
        "v_tanh_f32 %1, %1\n\t"
        "v_tanh_f32 %2, %2\n\t"
        "v_tanh_f32 %3, %3\n\t"
        "v_tanh_f32 %4, %4\n\t"
        "v_tanh_f32 %5, %5\n\t"
        "v_tanh_f32 %6, %6\n\t"
        "v_tanh_f32 %7, %7\n\t"
        "v_nop"
        : "+v"(a0), "+v"(a1), "+v"(a2), "+v"(a3),
          "+v"(a4), "+v"(a5), "+v"(a6), "+v"(a7));
    v[0] = a0; v[1] = a1; v[2] = a2; v[3] = a3;
    v[4] = a4; v[5] = a5; v[6] = a6; v[7] = a7;
}

// async-copy one contiguous 1856B x-tile (16 rows x 29 f32) into LDS.
// 116 16-byte chunks; excess lanes duplicate chunk 115 (same data, same addr)
// so EXEC stays full / branchless.
__device__ __forceinline__ void issue_x_copy(const float* gbase, const float* lbase, int lane) {
#pragma unroll
    for (int i = 0; i < 4; ++i) {
        int chunk = lane + 32 * i;
        if (i == 3 && chunk > 115) chunk = 115;
        unsigned lofs  = (unsigned)(uintptr_t)lbase + (unsigned)(chunk * 16);
        uint64_t gaddr = (uint64_t)(uintptr_t)gbase + (uint64_t)(chunk * 16);
        asm volatile("global_load_async_to_lds_b128 %0, %1, off"
                     :: "v"(lofs), "v"(gaddr) : "memory");
    }
}

__global__ __launch_bounds__(256) void rnn_wmma_kernel(
    const float* __restrict__ x,        // [T, B, V]
    const float* __restrict__ img_emb,  // [B, H]
    const float* __restrict__ W_ih,     // [H, V]
    const float* __restrict__ b_ih,     // [H]
    const float* __restrict__ W_hh,     // [H, H]
    const float* __restrict__ b_hh,     // [H]
    const float* __restrict__ W_ho,     // [V, H]
    const float* __restrict__ b_ho,     // [V]
    float* __restrict__ out)            // [T, B, V]
{
    // double-buffered x stage (doubles as coalesced output stage) + f16 h slab
    __shared__ __align__(16) float    xstage[2][WAVES_PER_BLOCK][XSTRIDE];
    __shared__ __align__(16) _Float16 lds_h[WAVES_PER_BLOCK][ROWS_PER_WAVE][32];

    const int lane = threadIdx.x & 31;
    const int wave = threadIdx.x >> 5;
    const int b0   = (blockIdx.x * WAVES_PER_BLOCK + wave) * ROWS_PER_WAVE;

    const int  c      = lane & 15;          // in-tile column (B/C layouts)
    const bool hiHalf = lane >= 16;
    const int  kb0    = hiHalf ? 8  : 0;    // A-fragment K base, elems 0..7
    const int  kb1    = hiHalf ? 24 : 16;   // A-fragment K base, elems 8..15
    const int  kB     = hiHalf ? 16 : 0;    // unused; B fragments use kb0/kb1 like A
    const int  mhi    = hiHalf ? 8  : 0;    // C-fragment row base
    const int  row    = lane & 15;          // A-fragment row

    const int n0 = c;        // tile-0 column index (hidden or vocab)
    const int n1 = 16 + c;   // tile-1 column index

    // ---- loop-invariant weight B-fragments: B[k][n] = W[n][k], zero-padded
    v16h Bi0, Bi1, Bh0, Bh1, Bo0, Bo1;
#pragma unroll
    for (int j = 0; j < 16; ++j) {
        const int k = (j < 8) ? (kb0 + j) : (kb1 + j - 8);
        Bi0[j] = (k < VOC)             ? (_Float16)W_ih[n0 * VOC + k] : (_Float16)0.f;
        Bi1[j] = (k < VOC && n1 < HID) ? (_Float16)W_ih[n1 * VOC + k] : (_Float16)0.f;
        Bh0[j] = (k < HID)             ? (_Float16)W_hh[n0 * HID + k] : (_Float16)0.f;
        Bh1[j] = (k < HID && n1 < HID) ? (_Float16)W_hh[n1 * HID + k] : (_Float16)0.f;
        Bo0[j] = (k < HID)             ? (_Float16)W_ho[n0 * HID + k] : (_Float16)0.f;
        Bo1[j] = (k < HID && n1 < VOC) ? (_Float16)W_ho[n1 * HID + k] : (_Float16)0.f;
    }

    // ---- per-lane bias scalars (zero on pad columns)
    const float bias0 = b_ih[n0] + b_hh[n0];                       // n0 < 20 always
    const float bias1 = (n1 < HID) ? (b_ih[n1] + b_hh[n1]) : 0.f;
    const float bo0v  = b_ho[n0];                                  // n0 < 16 < 29
    const float bo1v  = (n1 < VOC) ? b_ho[n1] : 0.f;

    // ---- img_emb as C-layout fragments (used only by the peeled t=0 step)
    v8f img0, img1;
#pragma unroll
    for (int r = 0; r < 8; ++r) {
        const int m = b0 + mhi + r;
        img0[r] = img_emb[m * HID + n0];
        img1[r] = (n1 < HID) ? img_emb[m * HID + n1] : 0.f;
    }

    // ---- kick off async copy of x tile for t = 0
    issue_x_copy(x + (size_t)b0 * VOC, &xstage[0][wave][0], lane);

    v16h hA = {};
    const bool valid1 = (n1 < VOC);

    auto rnn_step = [&](int t, bool first) {
        // ---- prefetch next x tile (double buffer), then wait for current one
        if (t + 1 < T_SEQ) {
            issue_x_copy(x + ((size_t)(t + 1) * NBATCH + (size_t)b0) * VOC,
                         &xstage[(t + 1) & 1][wave][0], lane);
            WAIT_ASYNC(4);
        } else {
            WAIT_ASYNC(0);
        }

        float* stage = &xstage[t & 1][wave][0];

        // ---- build x A-fragment from LDS stage (branchless pad masking)
        const float* xr = stage + row * VOC;
        v16h xA;
#pragma unroll
        for (int j = 0; j < 8; ++j) {
            const int k1 = kb1 + j;
            float a = xr[kb0 + j];                 // k <= 15, always valid
            float b = xr[k1];                      // in-bounds of padded stage
            b = (k1 < VOC) ? b : 0.f;
            xA[j]     = (_Float16)a;
            xA[8 + j] = (_Float16)b;
        }

        // ---- pre-activation: x@Wi^T (+ h@Wh^T, or img_emb on peeled first step)
        v8f acc0 = {}, acc1 = {};
        acc0 = __builtin_amdgcn_wmma_f32_16x16x32_f16(false, xA, false, Bi0, (short)0, acc0, false, false);
        acc1 = __builtin_amdgcn_wmma_f32_16x16x32_f16(false, xA, false, Bi1, (short)0, acc1, false, false);
        if (first) {
            acc0 += img0;
            acc1 += img1;
        } else {
            acc0 = __builtin_amdgcn_wmma_f32_16x16x32_f16(false, hA, false, Bh0, (short)0, acc0, false, false);
            acc1 = __builtin_amdgcn_wmma_f32_16x16x32_f16(false, hA, false, Bh1, (short)0, acc1, false, false);
        }

        // ---- h = tanh(acc + bias): hardware v_tanh_f32 (pad cols stay exactly 0)
        acc0 += bias0;
        acc1 += bias1;
        hw_tanh8(acc0);
        hw_tanh8(acc1);

        // ---- C-layout -> A-layout via wave-private LDS (pad cols write exact 0)
#pragma unroll
        for (int r = 0; r < 8; ++r) {
            lds_h[wave][mhi + r][n0] = (_Float16)acc0[r];
            lds_h[wave][mhi + r][n1] = (_Float16)acc1[r];
        }
        asm volatile("s_wait_dscnt 0" ::: "memory");
        {
            v8h lo = *(const v8h*)&lds_h[wave][row][kb0];
            v8h hi = *(const v8h*)&lds_h[wave][row][kb1];
#pragma unroll
            for (int j = 0; j < 8; ++j) { hA[j] = lo[j]; hA[8 + j] = hi[j]; }
        }

        // ---- logits = h @ W_ho^T + b_ho
        v8f l0 = {}, l1 = {};
        l0 = __builtin_amdgcn_wmma_f32_16x16x32_f16(false, hA, false, Bo0, (short)0, l0, false, false);
        l1 = __builtin_amdgcn_wmma_f32_16x16x32_f16(false, hA, false, Bo1, (short)0, l1, false, false);

        // ---- log_softmax over V=29, no max-pass needed: |h|<=1 bounds |logit|,
        //      exp can't overflow f32. Results staged to LDS (invalid v -> dump slot).
#pragma unroll
        for (int r = 0; r < 8; ++r) {
            const float v0 = l0[r] + bo0v;
            const float v1 = l1[r] + bo1v;          // exact 0 on invalid lanes
            float e = __expf(v0) + (valid1 ? __expf(v1) : 0.f);
#pragma unroll
            for (int s = 1; s <= 8; s <<= 1) e += __shfl_xor(e, s, 32);
            const float lse = __logf(e);
            const int m = mhi + r;
            stage[m * VOC + n0] = v0 - lse;
            const int idx1 = valid1 ? (m * VOC + n1) : (464 + c);  // branchless divert
            stage[idx1] = v1 - lse;
        }
        asm volatile("s_wait_dscnt 0" ::: "memory");

        // ---- coalesced flush: 116 x 16B chunks -> 4 global_store_b128 per lane
        float* ob = out + ((size_t)t * NBATCH + (size_t)b0) * VOC;
#pragma unroll
        for (int i = 0; i < 4; ++i) {
            int chunk = lane + 32 * i;
            if (i == 3 && chunk > 115) chunk = 115;   // duplicate store, same data
            v4f val = *(const v4f*)(stage + chunk * 4);
            *(v4f*)(ob + chunk * 4) = val;
        }
    };

    rnn_step(0, true);                                 // peeled: img_emb, no h@Wh
    for (int t = 1; t < T_SEQ; ++t) rnn_step(t, false);
}

extern "C" void kernel_launch(void* const* d_in, const int* in_sizes, int n_in,
                              void* d_out, int out_size, void* d_ws, size_t ws_size,
                              hipStream_t stream) {
    const float* x    = (const float*)d_in[0];
    const float* img  = (const float*)d_in[1];
    const float* W_ih = (const float*)d_in[2];
    const float* b_ih = (const float*)d_in[3];
    const float* W_hh = (const float*)d_in[4];
    const float* b_hh = (const float*)d_in[5];
    const float* W_ho = (const float*)d_in[6];
    const float* b_ho = (const float*)d_in[7];
    float* out = (float*)d_out;

    dim3 grid(NBATCH / (WAVES_PER_BLOCK * ROWS_PER_WAVE));  // 128 blocks
    dim3 block(WAVES_PER_BLOCK * 32);                        // 256 threads = 8 waves
    hipLaunchKernelGGL(rnn_wmma_kernel, grid, block, 0, stream,
                       x, img, W_ih, b_ih, W_hh, b_hh, W_ho, b_ho, out);
}
